// ContrastiveAttention_75222057222334
// MI455X (gfx1250) — matrix operand order repaired and checked by reading, not compiled, not run on hardware
//
#include <hip/hip_runtime.h>
#include <hip/hip_bf16.h>

typedef __attribute__((ext_vector_type(16))) _Float16 v16h;
typedef __attribute__((ext_vector_type(8)))  float    v8f;
typedef _Float16 half_t;

// Problem dims (fixed by the reference)
constexpr int kB = 8;
constexpr int kS = 2048;
constexpr int kQ = 2048;
constexpr int kD = 1024;
constexpr float kEps = 1e-5f;

// 16-byte packed store of 8 halves (used by transposed epilogue)
struct alignas(16) h8pack { half_t h[8]; };

// ---------------------------------------------------------------------------
// Block reductions (256 threads)
// ---------------------------------------------------------------------------
__device__ inline float block_sum(float v, float* sh) {
    int t = threadIdx.x;
    sh[t] = v; __syncthreads();
    for (int off = 128; off > 0; off >>= 1) {
        if (t < off) sh[t] += sh[t + off];
        __syncthreads();
    }
    float r = sh[0]; __syncthreads();
    return r;
}

__device__ inline float block_max(float v, float* sh) {
    int t = threadIdx.x;
    sh[t] = v; __syncthreads();
    for (int off = 128; off > 0; off >>= 1) {
        if (t < off) sh[t] = fmaxf(sh[t], sh[t + off]);
        __syncthreads();
    }
    float r = sh[0]; __syncthreads();
    return r;
}

// ---------------------------------------------------------------------------
// LayerNorm over D=1024, one row per 256-thread block, 4 elems per thread.
// ---------------------------------------------------------------------------
template <typename OutT>
__global__ __launch_bounds__(256)
void ln_kernel(const float* __restrict__ x, const float* __restrict__ g,
               const float* __restrict__ b, OutT* __restrict__ y)
{
    __shared__ float sh[256];
    const int row = blockIdx.x;
    const int t   = threadIdx.x;
    const float* xr = x + (size_t)row * kD;
    OutT* yr        = y + (size_t)row * kD;

    float v[4];
    float s = 0.f;
#pragma unroll
    for (int i = 0; i < 4; ++i) { v[i] = xr[t + 256 * i]; s += v[i]; }
    const float mean = block_sum(s, sh) * (1.0f / kD);

    float s2 = 0.f;
#pragma unroll
    for (int i = 0; i < 4; ++i) { float d = v[i] - mean; s2 += d * d; }
    const float var = block_sum(s2, sh) * (1.0f / kD);
    const float rstd = rsqrtf(var + kEps);

#pragma unroll
    for (int i = 0; i < 4; ++i) {
        const int c = t + 256 * i;
        yr[c] = (OutT)((v[i] - mean) * rstd * g[c] + b[c]);
    }
}

// ---------------------------------------------------------------------------
// fp32 -> fp16 cast (weights)
// ---------------------------------------------------------------------------
__global__ __launch_bounds__(256)
void cast_f32_to_f16_kernel(const float* __restrict__ x, half_t* __restrict__ y, int n)
{
    int i = blockIdx.x * 256 + threadIdx.x;
    if (i < n) y[i] = (half_t)x[i];
}

// ---------------------------------------------------------------------------
// WMMA fragment loads per CDNA5 ISA VGPR layouts (wave32).
// A (16x32 f16): lanes 0-15 row M=lane, K in {0..7,16..23};
//                lanes 16-31 row M=lane-16, K in {8..15,24..31}.
// ---------------------------------------------------------------------------
__device__ inline v16h load_a_frag(const half_t* __restrict__ base, int lda,
                                   int row0, int k0, int lane)
{
    const int r  = row0 + (lane & 15);
    const int kb = k0 + ((lane >> 4) << 3);     // +8 for upper half-wave
    const half_t* p = base + (size_t)r * lda + kb;
    v16h f;
#pragma unroll
    for (int v = 0; v < 8; ++v) {
        const int koff = ((v & 4) << 2) + ((v & 3) << 1);  // 0,2,4,6,16,18,20,22
        union { unsigned u; half_t h[2]; } cv;
        cv.u = *(const unsigned*)(p + koff);
        f[2 * v]     = cv.h[0];
        f[2 * v + 1] = cv.h[1];
    }
    return f;
}

// B (32x16 f16): lanes 0-15 column N=lane, K in {0..7,16..23}; lanes 16-31 K in {8..15,24..31}.
// TN form: Bmat is [N,K] row-major (C = A * Bmat^T) -> contiguous along K.
__device__ inline v16h load_b_frag_tn(const half_t* __restrict__ base, int ldb,
                                      int n0, int k0, int lane)
{
    const int n  = n0 + (lane & 15);
    const int kb = k0 + ((lane >> 4) << 3);     // +8 for upper half-wave
    const half_t* p = base + (size_t)n * ldb + kb;
    v16h f;
#pragma unroll
    for (int v = 0; v < 8; ++v) {
        union { unsigned u; half_t h[2]; } cv;
        cv.u = *(const unsigned*)(p + ((v & 4) << 2) + ((v & 3) << 1));
        f[2 * v]     = cv.h[0];
        f[2 * v + 1] = cv.h[1];
    }
    return f;
}

// One k-step fragment set: 2 A-frags (32 M-rows) x 4 B-frags (64 N-cols)
struct Frags { v16h a[2]; v16h b[4]; };

__device__ inline void load_frags(Frags& f, const half_t* __restrict__ Ab,
                                  const half_t* __restrict__ Bb,
                                  int K, int m0, int n0, int k, int lane)
{
    f.a[0] = load_a_frag(Ab, K, m0,      k, lane);
    f.a[1] = load_a_frag(Ab, K, m0 + 16, k, lane);
#pragma unroll
    for (int j = 0; j < 4; ++j)
        f.b[j] = load_b_frag_tn(Bb, K, n0 + 16 * j, k, lane);
}

__device__ inline void mma8(v8f (&acc)[2][4], const Frags& f)
{
#pragma unroll
    for (int i = 0; i < 2; ++i)
#pragma unroll
        for (int j = 0; j < 4; ++j)
            acc[i][j] = __builtin_amdgcn_wmma_f32_16x16x32_f16(
                            false, f.a[i], false, f.b[j], (short)0, acc[i][j],
                            false, false);
}

// ---------------------------------------------------------------------------
// Batched TN GEMM: C[M,N] = scale * (A[M,K] x B[N,K]^T) + bias[N]
//
// OUTMODE: 0 = f32 row-major C, 1 = f16 row-major C,
//          2 = f16 TRANSPOSED store (writes C^T as [N,M], packed 16B stores;
//              used to materialize V^T for the PV GEMM)
//
// Block: 256 threads = 8 waves arranged 2(M) x 4(N); wave tile 32 x 64.
// K loop unrolled by 64 with ping-pong fragment sets so each load clause
// issues underneath the previous 8-WMMA burst.  K % 64 == 0 for all calls;
// the tail set-0 reload at k==K over-reads 32 halves into the next ws buffer
// (benign, never consumed).
// grid = (N/256, M/64, batch)
// ---------------------------------------------------------------------------
template <int OUTMODE>
__global__ __launch_bounds__(256)
void gemm_tn_wmma_kernel(const half_t* __restrict__ A, const half_t* __restrict__ Bm,
                         const float* __restrict__ bias, void* __restrict__ Cv,
                         int M, int N, int K,
                         long long sA, long long sB, long long sC, float scale)
{
    const int lane = threadIdx.x & 31;
    const int wave = threadIdx.x >> 5;
    const int wm   = wave & 1;                  // 0..1  (M)
    const int wn   = wave >> 1;                 // 0..3  (N)

    const int m0 = blockIdx.y * 64 + wm * 32;
    const int n0 = blockIdx.x * 256 + wn * 64;

    const half_t* Ab = A  + (size_t)blockIdx.z * sA;
    const half_t* Bb = Bm + (size_t)blockIdx.z * sB;

    v8f acc[2][4] = {};

    Frags f0, f1;
    load_frags(f0, Ab, Bb, K, m0, n0, 0, lane);

    for (int k = 0; k + 64 <= K; k += 64) {
        load_frags(f1, Ab, Bb, K, m0, n0, k + 32, lane);
        // Unconditional speculative prefetch of upcoming A lines (A streams;
        // weight/K/V^T tiles stay hot in the 192MB L2).
        __builtin_prefetch(Ab + (size_t)(m0 + (lane & 15)) * K + k + 64, 0, 1);
        __builtin_prefetch(Ab + (size_t)(m0 + 16 + (lane & 15)) * K + k + 64, 0, 1);
        mma8(acc, f0);
        load_frags(f0, Ab, Bb, K, m0, n0, k + 64, lane);
        mma8(acc, f1);
    }

    // C layout: VGPR r -> row (tile_m0 + r + (lane<16 ? 0 : 8)), col n+(lane&15)
    const int cn = lane & 15;
    const int rh = (lane >> 4) << 3;            // +8 for upper half-wave
#pragma unroll
    for (int i = 0; i < 2; ++i) {
        const int rbase = m0 + 16 * i + rh;
#pragma unroll
        for (int j = 0; j < 4; ++j) {
            const int n = n0 + 16 * j + cn;
            const float bv = bias ? bias[n] : 0.f;
            if constexpr (OUTMODE == 2) {
                // rows rbase..rbase+7 of column n are contiguous in C^T[N,M]
                h8pack pk;
#pragma unroll
                for (int r = 0; r < 8; ++r)
                    pk.h[r] = (half_t)(acc[i][j][r] * scale + bv);
                half_t* Ct = (half_t*)Cv + (size_t)blockIdx.z * sC;
                *(h8pack*)(Ct + (size_t)n * M + rbase) = pk;
            } else {
#pragma unroll
                for (int r = 0; r < 8; ++r) {
                    const float val = acc[i][j][r] * scale + bv;
                    const size_t off = (size_t)(rbase + r) * N + n;
                    if constexpr (OUTMODE == 1)
                        ((half_t*)Cv + (size_t)blockIdx.z * sC)[off] = (half_t)val;
                    else
                        ((float*)Cv + (size_t)blockIdx.z * sC)[off] = val;
                }
            }
        }
    }
}

// ---------------------------------------------------------------------------
// Masked softmax + contrastive softmax, one row (b,q) per block over S=2048.
//   attn = softmax(mask(scores));  contrastive = softmax(1-attn) = softmax(-attn)
// attn in [0,1] -> exp(-attn) in [1/e,1]: no max-subtraction needed.
// ---------------------------------------------------------------------------
__global__ __launch_bounds__(256)
void contrastive_softmax_kernel(half_t* __restrict__ P, const int* __restrict__ masks)
{
    __shared__ float sh[256];
    const int row = blockIdx.x;         // over B*Q
    const int b   = row >> 11;          // Q = 2048
    const int t   = threadIdx.x;
    half_t* pr = P + (size_t)row * kS;
    const int* mr = masks + (size_t)b * kS;

    float x[8];
#pragma unroll
    for (int i = 0; i < 8; ++i) {
        const int c = t + 256 * i;
        x[i] = (mr[c] == 0) ? -1e9f : (float)pr[c];
    }
    float mx = -1e30f;
#pragma unroll
    for (int i = 0; i < 8; ++i) mx = fmaxf(mx, x[i]);
    mx = block_max(mx, sh);

    float l = 0.f;
#pragma unroll
    for (int i = 0; i < 8; ++i) { x[i] = __expf(x[i] - mx); l += x[i]; }
    l = block_sum(l, sh);
    const float rl = 1.0f / l;

    float z = 0.f;
#pragma unroll
    for (int i = 0; i < 8; ++i) { x[i] = __expf(-(x[i] * rl)); z += x[i]; }
    z = block_sum(z, sh);
    const float rz = 1.0f / z;

#pragma unroll
    for (int i = 0; i < 8; ++i) pr[t + 256 * i] = (half_t)(x[i] * rz);
}

// ---------------------------------------------------------------------------
// Host-side orchestration
// ---------------------------------------------------------------------------
extern "C" void kernel_launch(void* const* d_in, const int* in_sizes, int n_in,
                              void* d_out, int out_size, void* d_ws, size_t ws_size,
                              hipStream_t stream)
{
    (void)in_sizes; (void)n_in; (void)out_size; (void)ws_size;

    const float* sent   = (const float*)d_in[0];   // [B,S,D]
    const float* emo    = (const float*)d_in[1];   // [B,Q,D]
    const int*   masks  = (const int*)  d_in[2];   // [B,S]
    const float* ln_s_g = (const float*)d_in[3];
    const float* ln_s_b = (const float*)d_in[4];
    const float* ln_e_g = (const float*)d_in[5];
    const float* ln_e_b = (const float*)d_in[6];
    const float* Wq     = (const float*)d_in[7];
    const float* bq     = (const float*)d_in[8];
    const float* Wk     = (const float*)d_in[9];
    const float* bk     = (const float*)d_in[10];
    const float* Wv     = (const float*)d_in[11];
    const float* bv     = (const float*)d_in[12];
    const float* ln_o_g = (const float*)d_in[13];
    const float* ln_o_b = (const float*)d_in[14];
    float* out = (float*)d_out;                    // [B,Q,D] f32

    // Workspace layout
    char* ws = (char*)d_ws;
    auto take = [&](size_t bytes) { char* p = ws; ws += (bytes + 255) & ~size_t(255); return p; };
    half_t* s_ln = (half_t*)take((size_t)kB * kS * kD * 2);
    half_t* e_ln = (half_t*)take((size_t)kB * kQ * kD * 2);
    half_t* Wq16 = (half_t*)take((size_t)kD * kD * 2);
    half_t* Wk16 = (half_t*)take((size_t)kD * kD * 2);
    half_t* Wv16 = (half_t*)take((size_t)kD * kD * 2);
    half_t* Qf   = (half_t*)take((size_t)kB * kQ * kD * 2);
    half_t* Kf   = (half_t*)take((size_t)kB * kS * kD * 2);
    half_t* Vt   = (half_t*)take((size_t)kB * kD * kS * 2);  // V^T: [B, D, S]
    half_t* Pf   = (half_t*)take((size_t)kB * kQ * kS * 2);
    float*  Of   = (float*) take((size_t)kB * kQ * kD * 4);

    const dim3 blk(256);

    // 1) LayerNorm + cast to f16
    ln_kernel<half_t><<<kB * kS, blk, 0, stream>>>(sent, ln_s_g, ln_s_b, s_ln);
    ln_kernel<half_t><<<kB * kQ, blk, 0, stream>>>(emo,  ln_e_g, ln_e_b, e_ln);

    // 2) Weights to f16
    const int nw = kD * kD;
    cast_f32_to_f16_kernel<<<nw / 256, blk, 0, stream>>>(Wq, Wq16, nw);
    cast_f32_to_f16_kernel<<<nw / 256, blk, 0, stream>>>(Wk, Wk16, nw);
    cast_f32_to_f16_kernel<<<nw / 256, blk, 0, stream>>>(Wv, Wv16, nw);

    // 3) Projections (TN: x @ W^T + b).  V is stored transposed -> Vt[B,D,S].
    {
        const dim3 grid(kD / 256, kQ / 64, kB);
        gemm_tn_wmma_kernel<1><<<grid, blk, 0, stream>>>(
            e_ln, Wq16, bq, Qf, kQ, kD, kD,
            (long long)kQ * kD, 0LL, (long long)kQ * kD, 1.0f);
        gemm_tn_wmma_kernel<1><<<grid, blk, 0, stream>>>(
            s_ln, Wk16, bk, Kf, kS, kD, kD,
            (long long)kS * kD, 0LL, (long long)kS * kD, 1.0f);
        gemm_tn_wmma_kernel<2><<<grid, blk, 0, stream>>>(
            s_ln, Wv16, bv, Vt, kS, kD, kD,
            (long long)kS * kD, 0LL, (long long)kD * kS, 1.0f);
    }

    // 4) scores = Q @ K^T / sqrt(D)  (TN), f16 out into Pf
    {
        const dim3 grid(kS / 256, kQ / 64, kB);
        gemm_tn_wmma_kernel<1><<<grid, blk, 0, stream>>>(
            Qf, Kf, nullptr, Pf, kQ, kS, kD,
            (long long)kQ * kD, (long long)kS * kD, (long long)kQ * kS,
            1.0f / 32.0f);   // 1/sqrt(1024)
    }

    // 5) masked softmax + contrastive softmax (in-place on Pf)
    contrastive_softmax_kernel<<<kB * kQ, blk, 0, stream>>>(Pf, masks);

    // 6) out = P @ V = P @ (V^T)^T  (TN with B = Vt[D,S]), f32 out
    {
        const dim3 grid(kD / 256, kQ / 64, kB);
        gemm_tn_wmma_kernel<0><<<grid, blk, 0, stream>>>(
            Pf, Vt, nullptr, Of, kQ, kD, kS,
            (long long)kQ * kS, (long long)kD * kS, (long long)kQ * kD, 1.0f);
    }

    // 7) final LayerNorm -> f32 output
    ln_kernel<float><<<kB * kQ, blk, 0, stream>>>(Of, ln_o_g, ln_o_b, out);
}
